// NRU_RBN_36515811950830
// MI455X (gfx1250) — matrix operand order repaired and checked by reading, compiled, not run on hardware
//
#include <hip/hip_runtime.h>

// ---------------------------------------------------------------------------
// Types / helpers
// ---------------------------------------------------------------------------
typedef __attribute__((ext_vector_type(16))) __bf16 v16bf;
typedef __attribute__((ext_vector_type(2)))  __bf16 v2bf;
typedef __attribute__((ext_vector_type(8)))  float  v8f;

// Pack two f32 into two bf16 (RNE) using native hardware converts.
__device__ __forceinline__ unsigned pk2bf(float lo, float hi) {
  v2bf p;
  p.x = (__bf16)lo;
  p.y = (__bf16)hi;
  return __builtin_bit_cast(unsigned, p);
}
__device__ __forceinline__ __bf16 bfbits(unsigned short u) {
  return __builtin_bit_cast(__bf16, u);
}

// ---------------------------------------------------------------------------
// Generic WMMA GEMM:  C[M,N] = A[M,K] @ B[K,N] + bias[N]
//  - 128 threads = 4 waves; block tile 64x64; each wave: 16 rows x 64 cols
//  - K tiled by 32 through LDS (f32 -> bf16 RNE)
//  - M, N multiples of 64; K multiple of 4 (true for all call sites here);
//    full k-tiles take a guard-free vectorized path, only the tail tile
//    (K%32 != 0) uses per-float4 guards.
// Fragment layouts per CDNA5 ISA 7.12.2 (wave32):
//  A 16x32 bf16: lane L -> row M=L&15, half g=L>>4; vgpr v: K = {2v+8g, +1}
//                (v<4) and {16+2(v-4)+8g, +1} (v>=4)
//  B 32x16 bf16: lane L -> col N=L&15, half g=L>>4; vgpr v: K same as A
//  C/D 16x16 f32: vgpr r -> M = r + 8*(L>>4), N = L&15
// ---------------------------------------------------------------------------
__global__ __launch_bounds__(128)
void wmma_gemm(const float* __restrict__ A, const float* __restrict__ Bw,
               const float* __restrict__ bias, float* __restrict__ C,
               int M, int N, int K) {
  __shared__ unsigned short As[64][40];   // row stride 80B (8B aligned)
  __shared__ unsigned short Bs[32][72];   // row stride 144B (8B aligned)
  (void)M;
  const int m0   = blockIdx.x * 64;
  const int n0   = blockIdx.y * 64;
  const int tid  = threadIdx.x;
  const int wave = tid >> 5;
  const int lane = tid & 31;
  const int lm   = lane & 15;
  const int lg   = lane >> 4;

  const v8f vz = {0.f, 0.f, 0.f, 0.f, 0.f, 0.f, 0.f, 0.f};
  v8f acc[4];
  acc[0] = vz; acc[1] = vz; acc[2] = vz; acc[3] = vz;

  for (int k0 = 0; k0 < K; k0 += 32) {
    __syncthreads();
    if (k0 + 32 <= K) {
      // ---- fast path: full 32-wide tile, no per-lane guards ----
#pragma unroll
      for (int i = 0; i < 4; ++i) {             // A: 64x32 f32 = 512 float4
        int flat = tid + 128 * i;
        int r = flat >> 3, c4 = flat & 7;       // 8 float4 per row
        const float4 v = *(const float4*)(A + (size_t)(m0 + r) * K + k0 + c4 * 4);
        uint2 p; p.x = pk2bf(v.x, v.y); p.y = pk2bf(v.z, v.w);
        *(uint2*)&As[r][c4 * 4] = p;
      }
#pragma unroll
      for (int i = 0; i < 4; ++i) {             // B: 32x64 f32 = 512 float4
        int flat = tid + 128 * i;
        int r = flat >> 4, c4 = flat & 15;      // 16 float4 per row
        const float4 v = *(const float4*)(Bw + (size_t)(k0 + r) * N + n0 + c4 * 4);
        uint2 p; p.x = pk2bf(v.x, v.y); p.y = pk2bf(v.z, v.w);
        *(uint2*)&Bs[r][c4 * 4] = p;
      }
      if (k0 + 32 < K) {  // speculative prefetch of next tiles (global_prefetch_b8)
        __builtin_prefetch(&A[(size_t)(m0 + (tid >> 1)) * K + k0 + 32], 0, 0);
        __builtin_prefetch(&Bw[(size_t)(k0 + 32 + (tid >> 2)) * N + n0], 0, 0);
      }
    } else {
      // ---- tail tile: zero-pad K remainder (K % 4 == 0 at all call sites) ----
#pragma unroll
      for (int i = 0; i < 4; ++i) {
        int flat = tid + 128 * i;
        int r = flat >> 3, c4 = flat & 7;
        float4 v = {0.f, 0.f, 0.f, 0.f};
        if (k0 + c4 * 4 < K)
          v = *(const float4*)(A + (size_t)(m0 + r) * K + k0 + c4 * 4);
        uint2 p; p.x = pk2bf(v.x, v.y); p.y = pk2bf(v.z, v.w);
        *(uint2*)&As[r][c4 * 4] = p;
      }
#pragma unroll
      for (int i = 0; i < 4; ++i) {
        int flat = tid + 128 * i;
        int r = flat >> 4, c4 = flat & 15;
        float4 v = {0.f, 0.f, 0.f, 0.f};
        if (k0 + r < K)
          v = *(const float4*)(Bw + (size_t)(k0 + r) * N + n0 + c4 * 4);
        uint2 p; p.x = pk2bf(v.x, v.y); p.y = pk2bf(v.z, v.w);
        *(uint2*)&Bs[r][c4 * 4] = p;
      }
    }
    __syncthreads();

    v16bf af;
    const int arow = wave * 16 + lm;
#pragma unroll
    for (int v = 0; v < 8; ++v) {
      int kk = (v < 4) ? (2 * v + 8 * lg) : (16 + 2 * (v - 4) + 8 * lg);
      af[2 * v]     = bfbits(As[arow][kk]);
      af[2 * v + 1] = bfbits(As[arow][kk + 1]);
    }
#pragma unroll
    for (int j = 0; j < 4; ++j) {
      v16bf bf;
#pragma unroll
      for (int v = 0; v < 8; ++v) {
        int kk = (v < 4) ? (2 * v + 8 * lg) : (16 + 2 * (v - 4) + 8 * lg);
        bf[2 * v]     = bfbits(Bs[kk][j * 16 + lm]);
        bf[2 * v + 1] = bfbits(Bs[kk + 1][j * 16 + lm]);
      }
      acc[j] = __builtin_amdgcn_wmma_f32_16x16x32_bf16(
          false, af, false, bf, (short)0, acc[j], false, false);
    }
  }

#pragma unroll
  for (int j = 0; j < 4; ++j) {
    int n = n0 + j * 16 + lm;
    float bv = bias ? bias[n] : 0.f;
#pragma unroll
    for (int r = 0; r < 8; ++r) {
      int m = m0 + wave * 16 + r + 8 * lg;
      C[(size_t)m * N + n] = acc[j][r] + bv;
    }
  }
}

// ---------------------------------------------------------------------------
// Row-wise LayerNorm (+ activation). act: 0 none, 1 leaky(0.2), 2 tanh, 3 relu
// One block per row, 256 threads. dst may alias src.
// ---------------------------------------------------------------------------
__global__ __launch_bounds__(256)
void ln_act(float* __restrict__ dst, const float* __restrict__ src,
            const float* __restrict__ s, const float* __restrict__ o,
            int N, int act) {
  const float* x = src + (size_t)blockIdx.x * N;
  float* y = dst + (size_t)blockIdx.x * N;
  __shared__ float rs[256], rq[256];
  float sum = 0.f, sq = 0.f;
  for (int i = threadIdx.x; i < N; i += 256) { float v = x[i]; sum += v; sq += v * v; }
  rs[threadIdx.x] = sum; rq[threadIdx.x] = sq; __syncthreads();
  for (int st = 128; st > 0; st >>= 1) {
    if (threadIdx.x < st) { rs[threadIdx.x] += rs[threadIdx.x + st]; rq[threadIdx.x] += rq[threadIdx.x + st]; }
    __syncthreads();
  }
  float mean = rs[0] / N;
  float var  = rq[0] / N - mean * mean;
  float inv  = rsqrtf(var + 1e-5f);
  for (int i = threadIdx.x; i < N; i += 256) {
    float v = (x[i] - mean) * inv * s[i] + o[i];
    if      (act == 1) v = (v > 0.f) ? v : 0.2f * v;
    else if (act == 2) v = tanhf(v);
    else if (act == 3) v = fmaxf(v, 0.f);
    y[i] = v;
  }
}

// ---------------------------------------------------------------------------
// Small-N dense layer: Y[M,N] = act(X[M,K] @ W[K,N] + b)
// act: 0 none, 1 relu, 2 sigmoid, 3 softplus
// ---------------------------------------------------------------------------
__global__ void smalldot(const float* __restrict__ X, const float* __restrict__ W,
                         const float* __restrict__ b, float* __restrict__ Y,
                         int M, int N, int K, int act) {
  int t = blockIdx.x * blockDim.x + threadIdx.x;
  if (t >= M * N) return;
  int m = t / N, n = t % N;
  float acc = b ? b[n] : 0.f;
  const float* xr = X + (size_t)m * K;
  for (int k = 0; k < K; ++k) acc += xr[k] * W[(size_t)k * N + n];
  if      (act == 1) acc = fmaxf(acc, 0.f);
  else if (act == 2) acc = 1.f / (1.f + __expf(-acc));
  else if (act == 3) acc = (acc > 20.f) ? acc : log1pf(__expf(acc));
  Y[t] = acc;
}

// ---------------------------------------------------------------------------
// Row-wise concat of up to 3 sources into dst (rows x (wa+wb+wc))
// ---------------------------------------------------------------------------
__global__ void concat3(float* __restrict__ dst, const float* __restrict__ a, int wa,
                        const float* __restrict__ b, int wb,
                        const float* __restrict__ c, int wc, int rows) {
  int W = wa + wb + wc;
  int t = blockIdx.x * blockDim.x + threadIdx.x;
  if (t >= rows * W) return;
  int r = t / W, col = t % W;
  float v;
  if (col < wa)            v = a[(size_t)r * wa + col];
  else if (col < wa + wb)  v = b[(size_t)r * wb + (col - wa)];
  else                     v = c[(size_t)r * wc + (col - wa - wb)];
  dst[t] = v;
}

__global__ void zero_f(float* p, int n) {
  int i = blockIdx.x * blockDim.x + threadIdx.x;
  if (i < n) p[i] = 0.f;
}

// xin(B,12) = [x_enc[:,t,:], x_mark_enc[:,t,:]]
__global__ void build_xin_enc(float* __restrict__ dst, const float* __restrict__ x_enc,
                              const float* __restrict__ x_mark, int t) {
  int i = blockIdx.x * blockDim.x + threadIdx.x;
  if (i >= 128 * 12) return;
  int b = i / 12, c = i % 12;
  dst[i] = (c < 7) ? x_enc[((size_t)b * 256 + t) * 7 + c]
                   : x_mark[((size_t)b * 256 + t) * 5 + (c - 7)];
}
// xin(B,12) = [mu_prev, x_mark_dec[:,128+td,:]]
__global__ void build_xin_dec(float* __restrict__ dst, const float* __restrict__ mu_prev,
                              const float* __restrict__ x_mark_dec, int td) {
  int i = blockIdx.x * blockDim.x + threadIdx.x;
  if (i >= 128 * 12) return;
  int b = i / 12, c = i % 12;
  dst[i] = (c < 7) ? mu_prev[b * 7 + c]
                   : x_mark_dec[((size_t)b * 192 + 128 + td) * 5 + (c - 7)];
}

// ---------------------------------------------------------------------------
// NRU memory update. Per batch b (block), 256 threads = memory columns.
// v[b,k,m] = relu(u1[i]*u2[j]), i = k*8 + m/32, j = m%32;
// p5-normalize over m; m_state += 0.25 * sum_k(alpha_k*va_n - beta_k*vb_n)
// ---------------------------------------------------------------------------
__global__ __launch_bounds__(256)
void mem_update(float* __restrict__ mstate, const float* __restrict__ ua,
                const float* __restrict__ ub, const float* __restrict__ alpha,
                const float* __restrict__ beta) {
  int b = blockIdx.x;
  int t = threadIdx.x;
  __shared__ float su[128];
  __shared__ float sal[4], sbe[4];
  __shared__ float red[256];
  __shared__ float norms[8];
  if (t < 64)        su[t] = ua[b * 64 + t];
  else if (t < 128)  su[t] = ub[b * 64 + (t - 64)];
  if (t < 4)         sal[t] = alpha[b * 4 + t];
  else if (t < 8)    sbe[t - 4] = beta[b * 4 + (t - 4)];
  __syncthreads();
  float va[4], vb[4];
#pragma unroll
  for (int k = 0; k < 4; ++k) {
    int i = k * 8 + (t >> 5);
    int j = t & 31;
    va[k] = fmaxf(su[i] * su[32 + j], 0.f);
    vb[k] = fmaxf(su[64 + i] * su[96 + j], 0.f);
  }
  for (int idx = 0; idx < 8; ++idx) {
    float v = (idx < 4) ? va[idx] : vb[idx - 4];
    float p = v * v; p = p * p * v;           // v^5 (v >= 0)
    red[t] = p; __syncthreads();
    for (int s = 128; s > 0; s >>= 1) {
      if (t < s) red[t] += red[t + s];
      __syncthreads();
    }
    if (t == 0) norms[idx] = fmaxf(powf(red[0], 0.2f), 1e-12f);
    __syncthreads();
  }
  float d = 0.f;
#pragma unroll
  for (int k = 0; k < 4; ++k)
    d += sal[k] * va[k] / norms[k] - sbe[k] * vb[k] / norms[k + 4];
  mstate[b * 256 + t] += 0.25f * d;
}

// resin(B,255,12) = [mu_seq[s] - x_enc[:,s+1,:], x_mark_enc[:,s+1,:]]
__global__ void build_resin(float* __restrict__ resin, const float* __restrict__ mu_seq,
                            const float* __restrict__ x_enc, const float* __restrict__ x_mark) {
  int i = blockIdx.x * blockDim.x + threadIdx.x;
  if (i >= 128 * 255 * 12) return;
  int b = i / (255 * 12);
  int rem = i % (255 * 12);
  int s = rem / 12, c = rem % 12;
  float v;
  if (c < 7) v = mu_seq[((size_t)s * 128 + b) * 7 + c] - x_enc[((size_t)b * 256 + s + 1) * 7 + c];
  else       v = x_mark[((size_t)b * 256 + s + 1) * 5 + (c - 7)];
  resin[i] = v;
}

// Circular conv1d: X (B,255,512), W (192,255,3), Y (B,192,512)
__global__ __launch_bounds__(512)
void conv_circ(const float* __restrict__ X, const float* __restrict__ W,
               const float* __restrict__ bias, float* __restrict__ Y) {
  int o = blockIdx.x;      // 0..191
  int b = blockIdx.y;      // 0..127
  int l = threadIdx.x;     // 0..511
  __shared__ float sw[255 * 3];
  for (int i = threadIdx.x; i < 765; i += 512) sw[i] = W[(size_t)o * 765 + i];
  __syncthreads();
  const float* xb = X + (size_t)b * 255 * 512;
  float acc = bias[o];
  int lm1 = (l + 511) & 511, lp1 = (l + 1) & 511;
  for (int c = 0; c < 255; ++c) {
    const float* xc = xb + (size_t)c * 512;
    acc += sw[c * 3] * xc[lm1] + sw[c * 3 + 1] * xc[l] + sw[c * 3 + 2] * xc[lp1];
  }
  Y[((size_t)b * 192 + o) * 512 + l] = acc;
}

// Scatter mu/sigma sequences into out rows [0,319) and [319,638)
__global__ void write_musig(float* __restrict__ out, const float* __restrict__ mu_seq,
                            const float* __restrict__ sg_seq) {
  int i = blockIdx.x * blockDim.x + threadIdx.x;
  if (i >= 128 * 319 * 7) return;
  int b = i / (319 * 7);
  int rem = i % (319 * 7);
  int t = rem / 7, j = rem % 7;
  size_t src = ((size_t)t * 128 + b) * 7 + j;
  out[((size_t)b * 1022 + t) * 7 + j]       = mu_seq[src];
  out[((size_t)b * 1022 + 319 + t) * 7 + j] = sg_seq[src];
}
// Scatter res/ressigma into out rows [638,830) and [830,1022)
__global__ void write_res(float* __restrict__ out, const float* __restrict__ res,
                          const float* __restrict__ ressig) {
  int i = blockIdx.x * blockDim.x + threadIdx.x;
  if (i >= 128 * 192 * 7) return;
  int b = i / (192 * 7);
  int rem = i % (192 * 7);
  int o = rem / 7, j = rem % 7;
  out[((size_t)b * 1022 + 638 + o) * 7 + j] = res[i];
  out[((size_t)b * 1022 + 830 + o) * 7 + j] = ressig[i];
}

// ---------------------------------------------------------------------------
// Host orchestration
// Param flattening assumption (JAX pytree: dicts by sorted key; setup_inputs
// dict in insertion order): d_in[0..3] = x_enc, x_mark_enc, x_dec, x_mark_dec;
// d_in[4..] = params leaves:
//   cells[i] (i=0..2), 12 leaves each, base 4+12*i:
//     +0 hm2alpha.b +1 hm2alpha.w +2 hm2beta.b +3 hm2beta.w
//     +4 hm2v_alpha.b +5 hm2v_alpha.w +6 hm2v_beta.b +7 hm2v_beta.w
//     +8 hmi2h.b +9 hmi2h.w +10 ln_h.o +11 ln_h.s
//   40 conv.b  41 conv.w  42 dec_out.b  43 dec_out.w
//   decvar[j]: 44+4j = lin.b, lin.w, ln.o, ln.s
//   encvar[j]: 56+4j = lin.b, lin.w, ln.o, ln.s
//   72 fc_mu.b  73 fc_mu.w
//   74 W_h2o  75 W_h2o_sigma  76 b_o  77 b_o_sigma
//   78 varout.b 79 varout.w 80 varoutsigma.b 81 varoutsigma.w
// ---------------------------------------------------------------------------
extern "C" void kernel_launch(void* const* d_in, const int* in_sizes, int n_in,
                              void* d_out, int out_size, void* d_ws, size_t ws_size,
                              hipStream_t stream) {
  (void)in_sizes; (void)n_in; (void)out_size; (void)ws_size;
  const float* x_enc      = (const float*)d_in[0];
  const float* x_mark_enc = (const float*)d_in[1];
  const float* x_mark_dec = (const float*)d_in[3];
  auto P = [&](int i) { return (const float*)d_in[i]; };
  float* out = (float*)d_out;

  // ---- workspace carve (floats) ----
  float* w = (float*)d_ws;
  size_t off = 0;
  auto carve = [&](size_t n) {
    float* p = w + off;
    off += (n + 63) & ~(size_t)63;
    return p;
  };
  float* hstate  = carve(3 * 128 * 512);      // per-layer h
  float* mstate  = carve(3 * 128 * 256);      // per-layer memory
  float* xin     = carve(128 * 12);
  float* cbuf    = carve(128 * 1280);
  float* hlin    = carve(128 * 512);
  float* hm      = carve(128 * 768);
  float* alpha   = carve(128 * 4);
  float* beta    = carve(128 * 4);
  float* ua      = carve(128 * 64);
  float* ub      = carve(128 * 64);
  float* mu_seq  = carve(320 * 128 * 7);      // time-major (t,b,7)
  float* sg_seq  = carve(320 * 128 * 7);
  float* resin   = carve(128 * 255 * 12);
  float* res_t   = carve(24576 * 7);
  float* ressg_t = carve(24576 * 7);
  float* bufA    = carve((size_t)32640 * 1024);  // 133.7 MB
  float* bufB    = carve((size_t)32640 * 2048);  // 267.4 MB

  // ---- init recurrent state ----
  zero_f<<<(3 * 128 * 512 + 255) / 256, 256, 0, stream>>>(hstate, 3 * 128 * 512);
  zero_f<<<(3 * 128 * 256 + 255) / 256, 256, 0, stream>>>(mstate, 3 * 128 * 256);

  // ---- recurrence: 256 encoder + 64 decoder steps ----
  for (int t = 0; t < 320; ++t) {
    if (t < 256)
      build_xin_enc<<<6, 256, 0, stream>>>(xin, x_enc, x_mark_enc, t);
    else
      build_xin_dec<<<6, 256, 0, stream>>>(xin, mu_seq + (size_t)(t - 1) * 128 * 7,
                                           x_mark_dec, t - 256);
    const float* layin = xin;
    int insz = 12;
    for (int L = 0; L < 3; ++L) {
      int cb = 4 + 12 * L;
      int K = insz + 512 + 256;
      float* hL = hstate + (size_t)L * 128 * 512;
      float* mL = mstate + (size_t)L * 128 * 256;
      // c = [x, h, m]; h = relu(LN(c @ Whmi2h + b))
      concat3<<<(128 * K + 255) / 256, 256, 0, stream>>>(cbuf, layin, insz, hL, 512, mL, 256, 128);
      wmma_gemm<<<dim3(2, 8), 128, 0, stream>>>(cbuf, P(cb + 9), P(cb + 8), hlin, 128, 512, K);
      ln_act<<<128, 256, 0, stream>>>(hL, hlin, P(cb + 11), P(cb + 10), 512, 3);
      // hm = [h, m]
      concat3<<<(128 * 768 + 255) / 256, 256, 0, stream>>>(hm, hL, 512, mL, 256, nullptr, 0, 128);
      // alpha/beta = relu(hm @ W + b), N=4
      smalldot<<<4, 128, 0, stream>>>(hm, P(cb + 1), P(cb + 0), alpha, 128, 4, 768, 1);
      smalldot<<<4, 128, 0, stream>>>(hm, P(cb + 3), P(cb + 2), beta, 128, 4, 768, 1);
      // u_alpha/u_beta = hm @ W + b, N=64 (WMMA)
      wmma_gemm<<<dim3(2, 1), 128, 0, stream>>>(hm, P(cb + 5), P(cb + 4), ua, 128, 64, 768);
      wmma_gemm<<<dim3(2, 1), 128, 0, stream>>>(hm, P(cb + 7), P(cb + 6), ub, 128, 64, 768);
      mem_update<<<128, 256, 0, stream>>>(mL, ua, ub, alpha, beta);
      layin = hL;
      insz = 512;
    }
    // head
    smalldot<<<7, 128, 0, stream>>>(hstate + 2 * 128 * 512, P(74), P(76),
                                    mu_seq + (size_t)t * 128 * 7, 128, 7, 512, 0);
    smalldot<<<7, 128, 0, stream>>>(hstate + 2 * 128 * 512, P(75), P(77),
                                    sg_seq + (size_t)t * 128 * 7, 128, 7, 512, 2);
  }

  // ---- residual encoder (M = 128*255 = 32640) ----
  build_resin<<<(128 * 255 * 12 + 255) / 256, 256, 0, stream>>>(resin, mu_seq, x_enc, x_mark_enc);
  wmma_gemm<<<dim3(510, 16), 128, 0, stream>>>(resin, P(57), P(56), bufA, 32640, 1024, 12);
  ln_act<<<32640, 256, 0, stream>>>(bufA, bufA, P(59), P(58), 1024, 1);
  wmma_gemm<<<dim3(510, 32), 128, 0, stream>>>(bufA, P(61), P(60), bufB, 32640, 2048, 1024);
  ln_act<<<32640, 256, 0, stream>>>(bufB, bufB, P(63), P(62), 2048, 1);
  wmma_gemm<<<dim3(510, 16), 128, 0, stream>>>(bufB, P(65), P(64), bufA, 32640, 1024, 2048);
  ln_act<<<32640, 256, 0, stream>>>(bufA, bufA, P(67), P(66), 1024, 1);
  wmma_gemm<<<dim3(510, 8), 128, 0, stream>>>(bufA, P(69), P(68), bufB, 32640, 512, 1024);
  ln_act<<<32640, 256, 0, stream>>>(bufB, bufB, P(71), P(70), 512, 2);  // tanh
  wmma_gemm<<<dim3(510, 8), 128, 0, stream>>>(bufB, P(73), P(72), bufA, 32640, 512, 512);

  // ---- circular conv: bufA (B,255,512) -> bufB (B,192,512) ----
  conv_circ<<<dim3(192, 128), 512, 0, stream>>>(bufA, P(41), P(40), bufB);

  // ---- residual decoder (M = 128*192 = 24576) ----
  wmma_gemm<<<dim3(384, 16), 128, 0, stream>>>(bufB, P(45), P(44), bufA, 24576, 1024, 512);
  ln_act<<<24576, 256, 0, stream>>>(bufA, bufA, P(47), P(46), 1024, 1);
  wmma_gemm<<<dim3(384, 32), 128, 0, stream>>>(bufA, P(49), P(48), bufB, 24576, 2048, 1024);
  ln_act<<<24576, 256, 0, stream>>>(bufB, bufB, P(51), P(50), 2048, 1);
  wmma_gemm<<<dim3(384, 16), 128, 0, stream>>>(bufB, P(53), P(52), bufA, 24576, 1024, 2048);
  ln_act<<<24576, 256, 0, stream>>>(bufA, bufA, P(55), P(54), 1024, 1);
  wmma_gemm<<<dim3(384, 8), 128, 0, stream>>>(bufA, P(43), P(42), bufB, 24576, 512, 1024);
  smalldot<<<(24576 * 7 + 255) / 256, 256, 0, stream>>>(bufB, P(79), P(78), res_t, 24576, 7, 512, 0);
  smalldot<<<(24576 * 7 + 255) / 256, 256, 0, stream>>>(bufB, P(81), P(80), ressg_t, 24576, 7, 512, 3);

  // ---- assemble output (B, 1022, 7) ----
  write_musig<<<(128 * 319 * 7 + 255) / 256, 256, 0, stream>>>(out, mu_seq, sg_seq);
  write_res<<<(128 * 192 * 7 + 255) / 256, 256, 0, stream>>>(out, res_t, ressg_t);
}
